// DecoderBlock_56736517980327
// MI455X (gfx1250) — matrix operand order repaired and checked by reading, not compiled, not run on hardware
//
#include <hip/hip_runtime.h>
#include <hip/hip_bf16.h>

// ---------------------------------------------------------------------------
// DecoderBlock for MI455X (gfx1250): bf16 WMMA GEMMs (v_wmma_f32_16x16x32_bf16),
// fp32 accumulate, fp32 softmax/LN. KNN branch rewritten algebraically:
//   concat([f-xq, xq]) @ W  ==  P[idx] - Qt + Qb,
//   P = x@W_top, Qt = xq@W_top, Qb = xq@W_bot + b
// which removes the 131072-row gathered GEMMs entirely (~140 GFLOP saved).
// GEMM: block tile 128x128, 8 waves (4x2) of 32x64 -> 8 WMMA per 32-K step,
// register-double-buffered global->LDS staging + global_prefetch_b8 lookahead.
// ---------------------------------------------------------------------------

typedef __attribute__((ext_vector_type(16))) __bf16 bf16x16;
typedef __attribute__((ext_vector_type(8)))  __bf16 bf16x8;
typedef __attribute__((ext_vector_type(8)))  float  f32x8;

#define B_   16
#define N_   1024
#define NV_  1024
#define D_   384
#define H_   6
#define HD_  64
#define K_   8
#define ROWS_ (B_ * N_)      // 16384

__device__ __forceinline__ f32x8 wmma_bf16(bf16x16 a, bf16x16 b, f32x8 c) {
  return __builtin_amdgcn_wmma_f32_16x16x32_bf16(false, a, false, b, (short)0, c,
                                                 false, false);
}
__device__ __forceinline__ bf16x16 cat8(bf16x8 a, bf16x8 b) {
  return __builtin_shufflevector(a, b, 0, 1, 2, 3, 4, 5, 6, 7,
                                 8, 9, 10, 11, 12, 13, 14, 15);
}

// ---------------------------------------------------------------------------
// Weight prep: fp32 W[Kd][Nd] -> bf16 Wt[Nd][Kd]  (transpose+convert)
// ---------------------------------------------------------------------------
__global__ void wt_kernel(const float* __restrict__ W, __bf16* __restrict__ Wt,
                          int Kd, int Nd) {
  int t = blockIdx.x * blockDim.x + threadIdx.x;
  if (t >= Kd * Nd) return;
  int k = t / Nd, n = t - k * Nd;
  Wt[(size_t)n * Kd + k] = (__bf16)W[t];
}

// ---------------------------------------------------------------------------
// LayerNorm over D=384, one wave per row, writes bf16
// ---------------------------------------------------------------------------
__global__ __launch_bounds__(256) void ln_kernel(const float* __restrict__ x,
                                                 const float* __restrict__ g,
                                                 const float* __restrict__ bb,
                                                 __bf16* __restrict__ obf,
                                                 int rows) {
  int wave = threadIdx.x >> 5;
  int lane = threadIdx.x & 31;
  int row = blockIdx.x * 8 + wave;
  if (row >= rows) return;
  const float* xr = x + (size_t)row * D_;
  float v[12];
  float s = 0.f;
#pragma unroll
  for (int i = 0; i < 12; ++i) { v[i] = xr[lane + 32 * i]; s += v[i]; }
#pragma unroll
  for (int o = 16; o >= 1; o >>= 1) s += __shfl_xor(s, o, 32);
  float mean = s * (1.0f / D_);
  float vs = 0.f;
#pragma unroll
  for (int i = 0; i < 12; ++i) { float d = v[i] - mean; vs += d * d; }
#pragma unroll
  for (int o = 16; o >= 1; o >>= 1) vs += __shfl_xor(vs, o, 32);
  float inv = rsqrtf(vs * (1.0f / D_) + 1e-5f);
  __bf16* orow = obf + (size_t)row * D_;
#pragma unroll
  for (int i = 0; i < 12; ++i) {
    int c = lane + 32 * i;
    orow[c] = (__bf16)((v[i] - mean) * inv * g[c] + bb[c]);
  }
}

// ---------------------------------------------------------------------------
// Tiled WMMA GEMM: C[M][Nd] = act(A[M][Kd](bf16) x W + bias + Cin)
// Wt is pre-transposed [Nd][Kd] bf16. Block tile 128x128, 8 waves (4Mx2N),
// wave tile 32x64 -> acc[2][4], 8 WMMA per 32-K step. Global->LDS staging is
// software-pipelined through registers (one tile lookahead) so the
// s_wait_loadcnt lands behind the WMMA phase, plus prefetch two tiles ahead.
// act: 0=none, 1=exact GELU. Cout (f32) and CoutBf (bf16) both optional.
// ---------------------------------------------------------------------------
__global__ __launch_bounds__(256) void gemm_kernel(
    const __bf16* __restrict__ A, const __bf16* __restrict__ Wt,
    const float* __restrict__ bias, const float* __restrict__ Cin,
    float* __restrict__ Cout, __bf16* __restrict__ CoutBf,
    int M, int Nd, int Kd, int act) {
  __shared__ __bf16 As[128][32] __attribute__((aligned(64)));
  __shared__ __bf16 Bs[128][32] __attribute__((aligned(64)));
  int tid = threadIdx.x;
  int lane = tid & 31, wave = tid >> 5;
  int lrow = lane & 15, lhi = lane >> 4;
  int m0 = blockIdx.y * 128;
  int n0 = blockIdx.x * 128;
  int wm = wave >> 1, wn = wave & 1;  // 4x2 wave grid; wave tile 32x64

  const f32x8 fz = {0, 0, 0, 0, 0, 0, 0, 0};
  f32x8 acc[2][4];
#pragma unroll
  for (int i = 0; i < 2; ++i)
#pragma unroll
    for (int j = 0; j < 4; ++j) acc[i][j] = fz;

  int trow = tid >> 1, tseg = tid & 1;  // 128 rows x 2 segs of 16 bf16
  const __bf16* agBase = A + (size_t)(m0 + trow) * Kd + tseg * 16;
  const __bf16* bgBase = Wt + (size_t)(n0 + trow) * Kd + tseg * 16;

  // prologue: tile 0 into registers
  bf16x8 ra0 = *(const bf16x8*)agBase;
  bf16x8 ra1 = *(const bf16x8*)(agBase + 8);
  bf16x8 rb0 = *(const bf16x8*)bgBase;
  bf16x8 rb1 = *(const bf16x8*)(bgBase + 8);

  for (int k0 = 0; k0 < Kd; k0 += 32) {
    // commit held registers to LDS
    *(bf16x8*)&As[trow][tseg * 16] = ra0;
    *(bf16x8*)&As[trow][tseg * 16 + 8] = ra1;
    *(bf16x8*)&Bs[trow][tseg * 16] = rb0;
    *(bf16x8*)&Bs[trow][tseg * 16 + 8] = rb1;
    if (k0 + 32 < Kd) {
      // issue next tile's loads now; their wait lands after the WMMA phase
      const __bf16* ag = agBase + k0 + 32;
      const __bf16* bg = bgBase + k0 + 32;
      ra0 = *(const bf16x8*)ag;
      ra1 = *(const bf16x8*)(ag + 8);
      rb0 = *(const bf16x8*)bg;
      rb1 = *(const bf16x8*)(bg + 8);
      __builtin_prefetch(ag + 64, 0, 1);  // two tiles ahead -> L2
      __builtin_prefetch(bg + 64, 0, 1);
    }
    __syncthreads();

    bf16x16 afr[2], bfr[4];
#pragma unroll
    for (int mi = 0; mi < 2; ++mi) {
      // A frag: lane row=lrow, K = e + lhi*16 (contiguous 16)
      const __bf16* ap = &As[wm * 32 + mi * 16 + lrow][lhi * 16];
      afr[mi] = cat8(*(const bf16x8*)ap, *(const bf16x8*)(ap + 8));
    }
#pragma unroll
    for (int ni = 0; ni < 4; ++ni) {
      // B frag: lane col=lrow, K = e + lhi*16 (contiguous 16)
      const __bf16* bp = &Bs[wn * 64 + ni * 16 + lrow][lhi * 16];
      bfr[ni] = cat8(*(const bf16x8*)bp, *(const bf16x8*)(bp + 8));
    }
#pragma unroll
    for (int mi = 0; mi < 2; ++mi)
#pragma unroll
      for (int ni = 0; ni < 4; ++ni)
        acc[mi][ni] = wmma_bf16(afr[mi], bfr[ni], acc[mi][ni]);
    __syncthreads();
  }

#pragma unroll
  for (int ni = 0; ni < 4; ++ni) {
    int c = n0 + wn * 64 + ni * 16 + lrow;
    float bv = bias ? bias[c] : 0.f;   // hoisted per-column bias
#pragma unroll
    for (int mi = 0; mi < 2; ++mi) {
#pragma unroll
      for (int j = 0; j < 8; ++j) {
        int r = m0 + wm * 32 + mi * 16 + j + lhi * 8;
        float vv = acc[mi][ni][j] + bv;
        if (Cin) vv += Cin[(size_t)r * Nd + c];
        if (act == 1) vv = 0.5f * vv * (1.0f + erff(vv * 0.70710678118f));
        if (Cout) Cout[(size_t)r * Nd + c] = vv;
        if (CoutBf) CoutBf[(size_t)r * Nd + c] = (__bf16)vv;
      }
    }
  }
}

// ---------------------------------------------------------------------------
// Flash attention (one (b,h) per blockIdx.y, 128 query rows per block).
// Q/K/V are fp32 with row stride RS; converted to bf16 while staging to LDS.
// Each wave owns a 16-row strip; S = Q K^T and O += P V via WMMA, online
// softmax in fp32. Output written as bf16 [B*Nq][D_] at column h*64.
// ---------------------------------------------------------------------------
__global__ __launch_bounds__(256) void attn_kernel(
    const float* __restrict__ Qp, const float* __restrict__ Kp,
    const float* __restrict__ Vp, __bf16* __restrict__ Out, int RS, int Nq,
    int Nkv) {
  __shared__ __bf16 Qs[8][16][64] __attribute__((aligned(64)));  // 16 KB
  __shared__ __bf16 Kt[32][64]    __attribute__((aligned(64)));  // 4 KB
  __shared__ __bf16 Vt[64][32]    __attribute__((aligned(64)));  // 4 KB (d-major)
  __shared__ __bf16 Ps[8][16][32] __attribute__((aligned(64)));  // 8 KB
  int tid = threadIdx.x, lane = tid & 31, wave = tid >> 5;
  int lrow = lane & 15, lhi = lane >> 4;
  int bh = blockIdx.y;
  int b = bh / H_, h = bh - b * H_;
  int q0 = blockIdx.x * 128;
  size_t qbase = (size_t)b * Nq * RS + (size_t)h * HD_;
  size_t kvbase = (size_t)b * Nkv * RS + (size_t)h * HD_;

  {  // stage this wave's 16 Q rows, converting fp32 -> bf16
    int r = lrow;
    int d0 = lhi * 32;
    const float* src = Qp + qbase + (size_t)(q0 + wave * 16 + r) * RS + d0;
#pragma unroll
    for (int i = 0; i < 32; ++i) Qs[wave][r][d0 + i] = (__bf16)src[i];
  }
  __syncthreads();

  bf16x16 qf[2];
#pragma unroll
  for (int dk = 0; dk < 2; ++dk) {
    const __bf16* qp = &Qs[wave][lrow][dk * 32 + lhi * 16];
    qf[dk] = cat8(*(const bf16x8*)qp, *(const bf16x8*)(qp + 8));
  }

  const f32x8 fz = {0, 0, 0, 0, 0, 0, 0, 0};
  float mprev[8], lsum[8];
  f32x8 o[4];
#pragma unroll
  for (int j = 0; j < 8; ++j) { mprev[j] = -3.0e38f; lsum[j] = 0.f; }
#pragma unroll
  for (int nd = 0; nd < 4; ++nd) o[nd] = fz;
  const float scale = 0.125f;  // 1/sqrt(HD)

  for (int kv0 = 0; kv0 < Nkv; kv0 += 32) {
    __syncthreads();  // all waves done reading previous Kt/Vt
    {                 // stage K and V (V transposed to d-major)
      int kr = tid >> 3, seg = tid & 7;
      const float* ks = Kp + kvbase + (size_t)(kv0 + kr) * RS + seg * 8;
      const float* vsr = Vp + kvbase + (size_t)(kv0 + kr) * RS + seg * 8;
#pragma unroll
      for (int i = 0; i < 8; ++i) Kt[kr][seg * 8 + i] = (__bf16)ks[i];
#pragma unroll
      for (int i = 0; i < 8; ++i) Vt[seg * 8 + i][kr] = (__bf16)vsr[i];
    }
    __syncthreads();

    // S = Q K^T : 16 x 32 (two 16-col fragments), Kdim = 64 in two steps
    f32x8 s0 = fz, s1 = fz;
#pragma unroll
    for (int dk = 0; dk < 2; ++dk) {
      const __bf16* kp0 = &Kt[lrow][dk * 32 + lhi * 16];
      const __bf16* kp1 = &Kt[16 + lrow][dk * 32 + lhi * 16];
      bf16x16 k0f = cat8(*(const bf16x8*)kp0, *(const bf16x8*)(kp0 + 8));
      bf16x16 k1f = cat8(*(const bf16x8*)kp1, *(const bf16x8*)(kp1 + 8));
      s0 = wmma_bf16(qf[dk], k0f, s0);
      s1 = wmma_bf16(qf[dk], k1f, s1);
    }

    // online softmax (row = j + lhi*8, cols striped across 16-lane halves)
#pragma unroll
    for (int j = 0; j < 8; ++j) {
      float a = s0[j] * scale, c = s1[j] * scale;
      float mx = fmaxf(a, c);
#pragma unroll
      for (int off = 8; off >= 1; off >>= 1) mx = fmaxf(mx, __shfl_xor(mx, off, 16));
      float mnew = fmaxf(mprev[j], mx);
      float al = __expf(mprev[j] - mnew);
      float p0 = __expf(a - mnew), p1 = __expf(c - mnew);
      float rs = p0 + p1;
#pragma unroll
      for (int off = 8; off >= 1; off >>= 1) rs += __shfl_xor(rs, off, 16);
      lsum[j] = lsum[j] * al + rs;
      mprev[j] = mnew;
#pragma unroll
      for (int nd = 0; nd < 4; ++nd) o[nd][j] *= al;
      int r = j + lhi * 8;
      Ps[wave][r][lrow] = (__bf16)p0;        // C-layout -> LDS
      Ps[wave][r][16 + lrow] = (__bf16)p1;
    }

    // O += P x V  (P re-fragmented from wave-local LDS as A-layout)
    const __bf16* pp = &Ps[wave][lrow][lhi * 16];
    bf16x16 pf = cat8(*(const bf16x8*)pp, *(const bf16x8*)(pp + 8));
#pragma unroll
    for (int nd = 0; nd < 4; ++nd) {
      const __bf16* vp = &Vt[nd * 16 + lrow][lhi * 16];
      bf16x16 vf = cat8(*(const bf16x8*)vp, *(const bf16x8*)(vp + 8));
      o[nd] = wmma_bf16(pf, vf, o[nd]);
    }
  }

#pragma unroll
  for (int nd = 0; nd < 4; ++nd) {
#pragma unroll
    for (int j = 0; j < 8; ++j) {
      int q = q0 + wave * 16 + j + lhi * 8;
      int c = h * HD_ + nd * 16 + lrow;
      Out[((size_t)b * Nq + q) * D_ + c] = (__bf16)(o[nd][j] / lsum[j]);
    }
  }
}

// ---------------------------------------------------------------------------
// KNN gather-max: out[g][c] = max_k leaky(P[idx[b,k,n]][c] + Qb[g][c]-Qt[g][c])
// ---------------------------------------------------------------------------
__global__ __launch_bounds__(384) void gmax_kernel(
    const float* __restrict__ P, const float* __restrict__ Qt,
    const float* __restrict__ Qb, const int* __restrict__ idx,
    __bf16* __restrict__ outbf) {
  int g = blockIdx.x;       // query point 0..16383
  int c = threadIdx.x;      // column 0..383
  int b = g / N_, n = g - b * N_;
  float base = Qb[(size_t)g * D_ + c] - Qt[(size_t)g * D_ + c];
  float acc = -3.0e38f;
#pragma unroll
  for (int k = 0; k < K_; ++k) {
    int row = idx[((size_t)b * K_ + k) * N_ + n];
    float v = P[(size_t)row * D_ + c] + base;
    v = v > 0.f ? v : 0.2f * v;
    acc = fmaxf(acc, v);
  }
  outbf[(size_t)g * D_ + c] = (__bf16)acc;
}

// ---------------------------------------------------------------------------
extern "C" void kernel_launch(void* const* d_in, const int* in_sizes, int n_in,
                              void* d_out, int out_size, void* d_ws,
                              size_t ws_size, hipStream_t stream) {
  (void)in_sizes; (void)n_in; (void)out_size; (void)ws_size;
  const float* q_in   = (const float*)d_in[0];
  const float* v_in   = (const float*)d_in[1];
  const int*   knni   = (const int*)d_in[2];
  const int*   cknni  = (const int*)d_in[3];
  const float* n1_g   = (const float*)d_in[4];
  const float* n1_b   = (const float*)d_in[5];
  const float* qkv_w  = (const float*)d_in[6];
  const float* sproj_w= (const float*)d_in[7];
  const float* sproj_b= (const float*)d_in[8];
  const float* nq_g   = (const float*)d_in[9];
  const float* nq_b   = (const float*)d_in[10];
  const float* nv_g   = (const float*)d_in[11];
  const float* nv_b   = (const float*)d_in[12];
  const float* q_w    = (const float*)d_in[13];
  const float* k_w    = (const float*)d_in[14];
  const float* v_w    = (const float*)d_in[15];
  const float* cproj_w= (const float*)d_in[16];
  const float* cproj_b= (const float*)d_in[17];
  const float* knn_w  = (const float*)d_in[18];
  const float* knn_b  = (const float*)d_in[19];
  const float* merge_w= (const float*)d_in[20];
  const float* merge_b= (const float*)d_in[21];
  const float* cknn_w = (const float*)d_in[22];
  const float* cknn_b = (const float*)d_in[23];
  const float* cmerge_w=(const float*)d_in[24];
  const float* cmerge_b=(const float*)d_in[25];
  const float* n2_g   = (const float*)d_in[26];
  const float* n2_b   = (const float*)d_in[27];
  const float* fc1_w  = (const float*)d_in[28];
  const float* fc1_b  = (const float*)d_in[29];
  const float* fc2_w  = (const float*)d_in[30];
  const float* fc2_b  = (const float*)d_in[31];

  const size_t R = (size_t)ROWS_ * D_;     // 6,291,456 elements
  char* ws = (char*)d_ws;
  size_t off = 0;
  auto alloc = [&](size_t bytes) -> void* {
    void* p = ws + off;
    off += (bytes + 255) & ~(size_t)255;
    return p;
  };

  // bf16 transposed weights
  const size_t WSQ = (size_t)D_ * D_;      // 147456
  __bf16* wt_qkv   = (__bf16*)alloc((size_t)D_ * 3 * D_ * 2);
  __bf16* wt_sproj = (__bf16*)alloc(WSQ * 2);
  __bf16* wt_qw    = (__bf16*)alloc(WSQ * 2);
  __bf16* wt_kw    = (__bf16*)alloc(WSQ * 2);
  __bf16* wt_vw    = (__bf16*)alloc(WSQ * 2);
  __bf16* wt_cproj = (__bf16*)alloc(WSQ * 2);
  __bf16* wt_knn_t = (__bf16*)alloc(WSQ * 2);
  __bf16* wt_knn_b = (__bf16*)alloc(WSQ * 2);
  __bf16* wt_mrg_t = (__bf16*)alloc(WSQ * 2);
  __bf16* wt_mrg_b = (__bf16*)alloc(WSQ * 2);
  __bf16* wt_cknn_t= (__bf16*)alloc(WSQ * 2);
  __bf16* wt_cknn_b= (__bf16*)alloc(WSQ * 2);
  __bf16* wt_cmrg_t= (__bf16*)alloc(WSQ * 2);
  __bf16* wt_cmrg_b= (__bf16*)alloc(WSQ * 2);
  __bf16* wt_fc1   = (__bf16*)alloc((size_t)D_ * 4 * D_ * 2);
  __bf16* wt_fc2   = (__bf16*)alloc((size_t)4 * D_ * D_ * 2);

  // activations (with reuse)
  float*  big      = (float*)alloc((size_t)ROWS_ * 3 * D_ * 4);  // qkv / qc,kc,vc / P2,Qt2,Qb2 / h1_bf
  __bf16* normq_bf = (__bf16*)alloc(R * 2);                      // later: hn_bf
  __bf16* attn_bf  = (__bf16*)alloc(R * 2);                      // self & cross
  __bf16* x1_bf    = (__bf16*)alloc(R * 2);                      // q1_bf, later ca_bf
  float*  P1       = (float*)alloc(R * 4);                       // later qB
  float*  Qb1      = (float*)alloc(R * 4);
  __bf16* kf_bf    = (__bf16*)alloc(R * 2);                      // kf / kf2
  float*  tmpf     = (float*)alloc(R * 4);
  float*  qA       = (float*)alloc(R * 4);
  __bf16* nq_bf    = (__bf16*)alloc(R * 2);
  __bf16* nv_bf    = (__bf16*)alloc(R * 2);

  float* qc = big;
  float* kc = big + R;
  float* vc = big + 2 * R;
  float* P2 = big;
  float* Qt2 = big + R;
  float* Qb2 = big + 2 * R;
  __bf16* h1_bf = (__bf16*)big;
  float* qB = P1;
  __bf16* hn_bf = normq_bf;
  __bf16* ca_bf = x1_bf;
  __bf16* kf2_bf = kf_bf;

  auto wt = [&](const float* W, __bf16* Wt, int Kd, int Nd) {
    int tot = Kd * Nd;
    wt_kernel<<<dim3((tot + 255) / 256), dim3(256), 0, stream>>>(W, Wt, Kd, Nd);
  };
  auto gemm = [&](const __bf16* A, const __bf16* Wtp, const float* bias,
                  const float* Cin, float* Cout, __bf16* CoutBf, int M, int Nd,
                  int Kd, int act) {
    gemm_kernel<<<dim3(Nd / 128, M / 128), dim3(256), 0, stream>>>(
        A, Wtp, bias, Cin, Cout, CoutBf, M, Nd, Kd, act);
  };
  auto ln = [&](const float* x, const float* g, const float* b, __bf16* o) {
    ln_kernel<<<dim3(ROWS_ / 8), dim3(256), 0, stream>>>(x, g, b, o, ROWS_);
  };

  // ---- weight prep ----
  wt(qkv_w, wt_qkv, D_, 3 * D_);
  wt(sproj_w, wt_sproj, D_, D_);
  wt(q_w, wt_qw, D_, D_);
  wt(k_w, wt_kw, D_, D_);
  wt(v_w, wt_vw, D_, D_);
  wt(cproj_w, wt_cproj, D_, D_);
  wt(knn_w, wt_knn_t, D_, D_);
  wt(knn_w + WSQ, wt_knn_b, D_, D_);
  wt(merge_w, wt_mrg_t, D_, D_);
  wt(merge_w + WSQ, wt_mrg_b, D_, D_);
  wt(cknn_w, wt_cknn_t, D_, D_);
  wt(cknn_w + WSQ, wt_cknn_b, D_, D_);
  wt(cmerge_w, wt_cmrg_t, D_, D_);
  wt(cmerge_w + WSQ, wt_cmrg_b, D_, D_);
  wt(fc1_w, wt_fc1, D_, 4 * D_);
  wt(fc2_w, wt_fc2, 4 * D_, D_);

  // ---- stage 1: self attention + knn + merge ----
  ln(q_in, n1_g, n1_b, normq_bf);
  gemm(normq_bf, wt_qkv, nullptr, nullptr, big, nullptr, ROWS_, 3 * D_, D_, 0);
  attn_kernel<<<dim3(N_ / 128, B_ * H_), dim3(256), 0, stream>>>(
      big, big + D_, big + 2 * D_, attn_bf, 3 * D_, N_, N_);
  gemm(attn_bf, wt_sproj, sproj_b, nullptr, nullptr, x1_bf, ROWS_, D_, D_, 0);
  gemm(normq_bf, wt_knn_t, nullptr, nullptr, P1, nullptr, ROWS_, D_, D_, 0);
  gemm(normq_bf, wt_knn_b, knn_b, nullptr, Qb1, nullptr, ROWS_, D_, D_, 0);
  gmax_kernel<<<dim3(ROWS_), dim3(D_), 0, stream>>>(P1, P1, Qb1, knni, kf_bf);
  gemm(x1_bf, wt_mrg_t, nullptr, q_in, tmpf, nullptr, ROWS_, D_, D_, 0);
  gemm(kf_bf, wt_mrg_b, merge_b, tmpf, qA, nullptr, ROWS_, D_, D_, 0);

  // ---- stage 2: cross attention + cross knn + merge ----
  ln(qA, nq_g, nq_b, nq_bf);
  ln(v_in, nv_g, nv_b, nv_bf);
  gemm(nq_bf, wt_qw, nullptr, nullptr, qc, nullptr, ROWS_, D_, D_, 0);
  gemm(nv_bf, wt_kw, nullptr, nullptr, kc, nullptr, ROWS_, D_, D_, 0);
  gemm(nv_bf, wt_vw, nullptr, nullptr, vc, nullptr, ROWS_, D_, D_, 0);
  attn_kernel<<<dim3(N_ / 128, B_ * H_), dim3(256), 0, stream>>>(
      qc, kc, vc, attn_bf, D_, N_, NV_);
  gemm(attn_bf, wt_cproj, cproj_b, nullptr, nullptr, ca_bf, ROWS_, D_, D_, 0);
  gemm(nv_bf, wt_cknn_t, nullptr, nullptr, P2, nullptr, ROWS_, D_, D_, 0);
  gemm(nq_bf, wt_cknn_t, nullptr, nullptr, Qt2, nullptr, ROWS_, D_, D_, 0);
  gemm(nq_bf, wt_cknn_b, cknn_b, nullptr, Qb2, nullptr, ROWS_, D_, D_, 0);
  gmax_kernel<<<dim3(ROWS_), dim3(D_), 0, stream>>>(P2, Qt2, Qb2, cknni, kf2_bf);
  gemm(ca_bf, wt_cmrg_t, nullptr, qA, tmpf, nullptr, ROWS_, D_, D_, 0);
  gemm(kf2_bf, wt_cmrg_b, cmerge_b, tmpf, qB, nullptr, ROWS_, D_, D_, 0);

  // ---- stage 3: MLP ----
  ln(qB, n2_g, n2_b, hn_bf);
  gemm(hn_bf, wt_fc1, fc1_b, nullptr, nullptr, h1_bf, ROWS_, 4 * D_, D_, 1);
  gemm(h1_bf, wt_fc2, fc2_b, qB, (float*)d_out, nullptr, ROWS_, D_, 4 * D_, 0);
}
